// Switcher_16638703305121
// MI455X (gfx1250) — matrix operand order repaired and checked by reading, not compiled, not run on hardware
//
#include <hip/hip_runtime.h>

// ---------------------------------------------------------------------------
// Switcher, MI455X/gfx1250:
//   pre:   Wb -> bf16 (already n-major); W -> bf16, transposed to [lang][n][k]
//   GEMM1: h(bf16) = x(f32->bf16) @ Wb^T + b_base
//   GEMM2: out(f32) = h(bf16) @ Wt[id_b] + bias[id_b]
// Main loops: double-buffered LDS fed by global_load_async_to_lds_b128
// (ASYNCcnt / s_wait_asynccnt), compute on v_wmma_f32_16x16x32_bf16.
// ---------------------------------------------------------------------------

typedef __attribute__((ext_vector_type(16))) __bf16        v16bf;
typedef __attribute__((ext_vector_type(8)))  float         v8f;
typedef __attribute__((ext_vector_type(4)))  float         f4;
typedef __attribute__((ext_vector_type(4)))  int           i4;
typedef __attribute__((ext_vector_type(4)))  unsigned int  u4;
typedef __attribute__((ext_vector_type(2)))  unsigned int  u2;

#define SEQ  2048
#define BZ   16
#define DIM  1024
#define BM   128
#define BN   128
#define BK   64
#define LDT  72    // padded LDS row stride in bf16 elems (144B: 16B-aligned, conflict-free)

// ---------------- f32 -> bf16 via native fptrunc (v_cvt_pk_bf16_f32) --------
__device__ __forceinline__ unsigned int pk_bf16(float a, float b) {
    union { __bf16 h[2]; unsigned int u; } r;
    r.h[0] = (__bf16)a; r.h[1] = (__bf16)b;
    return r.u;
}
__device__ __forceinline__ unsigned short cvt1_bf16(float a) {
    union { __bf16 h; unsigned short u; } r;
    r.h = (__bf16)a;
    return r.u;
}

// ---------------- async global->LDS copy (gfx1250) --------------------------
#if __has_builtin(__builtin_amdgcn_global_load_async_to_lds_b128)
#define HAVE_ASYNC_LDS 1
#else
#define HAVE_ASYNC_LDS 0
#endif

typedef __attribute__((address_space(1))) i4* gas_i4p;
typedef __attribute__((address_space(3))) i4* las_i4p;

__device__ __forceinline__ void copy_b128_g2l(const unsigned short* g, unsigned short* l) {
#if HAVE_ASYNC_LDS
    __builtin_amdgcn_global_load_async_to_lds_b128((gas_i4p)g, (las_i4p)l, 0, 0);
#else
    *(u4*)l = *(const u4*)g;
#endif
}

template <int N>
__device__ __forceinline__ void async_wait_le() {
#if HAVE_ASYNC_LDS
#if __has_builtin(__builtin_amdgcn_s_wait_asynccnt)
    __builtin_amdgcn_s_wait_asynccnt(N);
#else
    asm volatile("s_wait_asynccnt %0" :: "i"(N) : "memory");
#endif
#endif
}

union Frag {
    u4    q[2];
    v16bf v;
};

// ---------------------------------------------------------------------------
// Tiled GEMM core. B is always bf16 n-major (B[n*DIM + k]).
// A_BF16: A is bf16 row-major (async-copied); else A is f32 (convert on stage).
// C_BF16: C stored as bf16; else f32.
// ---------------------------------------------------------------------------
template <bool A_BF16, bool C_BF16>
__device__ __forceinline__ void gemm_core(
    const void* Aptr, size_t lda, size_t aoff,      // elems
    const unsigned short* __restrict__ Bh,
    const float* __restrict__ bias,
    void* Cptr, size_t ldc, size_t coff,            // elems
    int rowBase, int nBase)
{
    __shared__ __align__(16) unsigned short smA[2][BM * LDT];
    __shared__ __align__(16) unsigned short smB[2][BN * LDT];

    const int tid  = threadIdx.x;
    const int lane = tid & 31;
    const int wid  = tid >> 5;            // 8 waves
    const int mw   = wid >> 1;            // 0..3  (M sub-tile)
    const int nw   = wid & 1;             // 0..1  (N sub-tile)
    const int l16  = lane & 15;
    const int kh   = (lane >> 4) * 8;     // A frag K-half (0 / 8)
    const int kb   = (lane >> 4) * 16;    // unused (B frag uses kh)

    v8f acc[2][4];
    #pragma unroll
    for (int i = 0; i < 2; ++i)
        #pragma unroll
        for (int j = 0; j < 4; ++j)
            #pragma unroll
            for (int r = 0; r < 8; ++r)
                acc[i][j][r] = 0.0f;

    auto stageA = [&](int buf, int k0) {
        if (A_BF16) {
            const unsigned short* Ah = (const unsigned short*)Aptr;
            #pragma unroll
            for (int it = 0; it < 4; ++it) {
                int chunk = tid + it * 256;          // 1024 chunks of 8 bf16
                int r  = chunk >> 3;                 // 0..127
                int c8 = (chunk & 7) * 8;            // 0..56
                copy_b128_g2l(Ah + aoff + (size_t)(rowBase + r) * lda + (size_t)(k0 + c8),
                              &smA[buf][r * LDT + c8]);
            }
        } else {
            const float* Af = (const float*)Aptr;
            #pragma unroll
            for (int it = 0; it < 8; ++it) {
                int chunk = tid + it * 256;          // 2048 chunks of 4 f32
                int r  = chunk >> 4;                 // 0..127
                int c4 = (chunk & 15) * 4;           // 0..60
                f4 v = *(const f4*)(Af + aoff + (size_t)(rowBase + r) * lda + (size_t)(k0 + c4));
                u2 p; p.x = pk_bf16(v.x, v.y); p.y = pk_bf16(v.z, v.w);
                *(u2*)(&smA[buf][r * LDT + c4]) = p;
            }
        }
    };
    auto stageB = [&](int buf, int k0) {
        #pragma unroll
        for (int it = 0; it < 4; ++it) {
            int chunk = tid + it * 256;
            int n  = chunk >> 3;
            int c8 = (chunk & 7) * 8;
            copy_b128_g2l(Bh + (size_t)(nBase + n) * DIM + (size_t)(k0 + c8),
                          &smB[buf][n * LDT + c8]);
        }
    };

    constexpr int NASY = (A_BF16 ? 8 : 4);   // async ops in flight for one tile
    constexpr int S    = DIM / BK;           // 16 steps

    stageA(0, 0);
    stageB(0, 0);

    for (int s = 0; s < S; ++s) {
        const int cur = s & 1;
        if (s + 1 < S) {                      // prefetch next tile into other buffer
            stageA(1 - cur, (s + 1) * BK);
            stageB(1 - cur, (s + 1) * BK);
            async_wait_le<NASY>();            // current tile's copies complete
        } else {
            async_wait_le<0>();
        }
        __syncthreads();

        #pragma unroll
        for (int kk = 0; kk < BK; kk += 32) {
            Frag af[2], bf[4];
            #pragma unroll
            for (int i = 0; i < 2; ++i) {
                // A 16-bit layout: lanes 0-15 K 0-7 & 16-23; lanes 16-31 K 8-15 & 24-31
                const u4* p = (const u4*)&smA[cur][(mw * 32 + i * 16 + l16) * LDT + kk + kh];
                af[i].q[0] = p[0];
                af[i].q[1] = p[2];            // +32B == +16 elems
            }
            #pragma unroll
            for (int j = 0; j < 4; ++j) {
                // B 16-bit layout: lanes 0-15 K 0-7 & 16-23; lanes 16-31 K 8-15 & 24-31
                const u4* p = (const u4*)&smB[cur][(nw * 64 + j * 16 + l16) * LDT + kk + kh];
                bf[j].q[0] = p[0];
                bf[j].q[1] = p[2];
            }
            #pragma unroll
            for (int i = 0; i < 2; ++i)
                #pragma unroll
                for (int j = 0; j < 4; ++j)
                    acc[i][j] = __builtin_amdgcn_wmma_f32_16x16x32_bf16(
                        false, af[i].v, false, bf[j].v,
                        (short)0, acc[i][j], false, false);
        }
        __syncthreads();                      // buffer may be overwritten next step
    }

    // ---- epilogue: bias add + store ----
    float bv[4];
    #pragma unroll
    for (int j = 0; j < 4; ++j)
        bv[j] = bias[nBase + nw * 64 + j * 16 + l16];

    const int rsub = (lane >> 4) * 8;
    #pragma unroll
    for (int i = 0; i < 2; ++i)
        #pragma unroll
        for (int j = 0; j < 4; ++j) {
            int col = nBase + nw * 64 + j * 16 + l16;
            #pragma unroll
            for (int r = 0; r < 8; ++r) {
                int row = rowBase + mw * 32 + i * 16 + rsub + r;
                if (C_BF16) {
                    unsigned short* Ch = (unsigned short*)Cptr;
                    Ch[coff + (size_t)row * ldc + (size_t)col] = cvt1_bf16(acc[i][j][r] + bv[j]);
                } else {
                    float* Cf = (float*)Cptr;
                    Cf[coff + (size_t)row * ldc + (size_t)col] = acc[i][j][r] + bv[j];
                }
            }
        }
}

// ---------------- pre-pass: W_base f32 -> bf16 (layout kept: [n][k]) --------
__global__ __launch_bounds__(256)
void cvt_wb(const float* __restrict__ src, unsigned short* __restrict__ dst)
{
    size_t i = ((size_t)blockIdx.x * 256 + threadIdx.x) * 4;
    f4 v = *(const f4*)(src + i);
    u2 p; p.x = pk_bf16(v.x, v.y); p.y = pk_bf16(v.z, v.w);
    *(u2*)(dst + i) = p;
}

// -------- pre-pass: W [lang][d][e] f32 -> Wt [lang][e][d] bf16 (LDS tiled) --
__global__ __launch_bounds__(256)
void cvt_transpose_w(const float* __restrict__ W, unsigned short* __restrict__ Wt)
{
    __shared__ __align__(16) unsigned short T[64 * LDT];   // 64x64 tile, stride 72

    const int lang  = blockIdx.z;
    const int dbase = blockIdx.x * 64;
    const int ebase = blockIdx.y * 64;
    const int tid   = threadIdx.x;

    const float* src = W + ((size_t)lang * DIM + dbase) * DIM + ebase;
    const int tr = tid >> 4;              // 0..15
    const int tc = (tid & 15) * 4;        // 0..60
    #pragma unroll
    for (int it = 0; it < 4; ++it) {
        int d = tr + it * 16;
        f4 v = *(const f4*)(src + (size_t)d * DIM + tc);
        T[(tc + 0) * LDT + d] = cvt1_bf16(v.x);
        T[(tc + 1) * LDT + d] = cvt1_bf16(v.y);
        T[(tc + 2) * LDT + d] = cvt1_bf16(v.z);
        T[(tc + 3) * LDT + d] = cvt1_bf16(v.w);
    }
    __syncthreads();

    unsigned short* dst = Wt + ((size_t)lang * DIM + ebase) * DIM + dbase;
    #pragma unroll
    for (int it = 0; it < 2; ++it) {
        int chunk = tid + it * 256;       // 512 chunks of 8 bf16
        int e  = chunk >> 3;              // 0..63
        int c8 = (chunk & 7) * 8;         // 0..56
        *(u4*)(dst + (size_t)e * DIM + c8) = *(const u4*)&T[e * LDT + c8];
    }
}

// ---------------- GEMM1: h(bf16) = x @ Wb^T + b_base ------------------------
__global__ __launch_bounds__(256, 2)
void switcher_gemm1(const float* __restrict__ x,
                    const unsigned short* __restrict__ Wbh,
                    const float* __restrict__ bb,
                    unsigned short* __restrict__ h)
{
    gemm_core<false, true>(x, DIM, 0, Wbh, bb, h, DIM, 0,
                           blockIdx.x * BM, blockIdx.y * BN);
}

// ---------------- GEMM2: out(f32) = h(bf16) @ Wt[id_b] + bias[id_b] ---------
__global__ __launch_bounds__(256, 2)
void switcher_gemm2(const unsigned short* __restrict__ h,
                    const unsigned short* __restrict__ Wt,
                    const float* __restrict__ biasAll,
                    const int*   __restrict__ lang_ids,
                    const int*   __restrict__ dict_len,
                    float*       __restrict__ out)
{
    const int b  = blockIdx.z;
    const int id = dict_len[0] - 1 - lang_ids[b];        // expert index

    gemm_core<true, false>(h, (size_t)BZ * DIM, (size_t)b * DIM,
                           Wt + (size_t)id * DIM * DIM,
                           biasAll + (size_t)id * DIM,
                           out, (size_t)BZ * DIM, (size_t)b * DIM,
                           blockIdx.x * BM, blockIdx.y * BN);
}

extern "C" void kernel_launch(void* const* d_in, const int* in_sizes, int n_in,
                              void* d_out, int out_size, void* d_ws, size_t ws_size,
                              hipStream_t stream)
{
    const float* x        = (const float*)d_in[0];
    const int*   lang_ids = (const int*)  d_in[1];
    const float* W_base   = (const float*)d_in[2];
    const float* b_base   = (const float*)d_in[3];
    const float* W        = (const float*)d_in[4];
    const float* bias     = (const float*)d_in[5];
    const int*   dict_len = (const int*)  d_in[6];
    (void)in_sizes; (void)n_in; (void)out_size; (void)ws_size;

    // workspace layout (bf16): Wb 2MB | Wt 32MB | h 64MB  (total 98MB)
    unsigned short* wbh = (unsigned short*)d_ws;
    unsigned short* wth = wbh + ((size_t)DIM * DIM);
    unsigned short* hh  = wth + ((size_t)16 * DIM * DIM);
    float* out = (float*)d_out;

    dim3 blk(256, 1, 1);

    cvt_wb<<<dim3((DIM * DIM) / (256 * 4)), blk, 0, stream>>>(W_base, wbh);
    cvt_transpose_w<<<dim3(DIM / 64, DIM / 64, 16), blk, 0, stream>>>(W, wth);

    switcher_gemm1<<<dim3((SEQ * BZ) / BM, DIM / BN), blk, 0, stream>>>(x, wbh, b_base, hh);
    switcher_gemm2<<<dim3(SEQ / BM, DIM / BN, BZ), blk, 0, stream>>>(hh, wth, bias, lang_ids,
                                                                     dict_len, out);
}
